// GNN_encoder_58196806860862
// MI455X (gfx1250) — matrix operand order repaired and checked by reading, not compiled, not run on hardware
//
#include <hip/hip_runtime.h>
#include <hip/hip_bf16.h>

typedef __attribute__((ext_vector_type(16))) _Float16 v16h;
typedef __attribute__((ext_vector_type(8)))  float    v8f;

#define N_NODES 10000
#define N_EDGES 320000
#define F_IN    5
#define F_OUT   64
#define KDIM    32          // MLP hidden = WMMA K
#define HCOL    320         // F_IN * F_OUT
#define C_GAT   27
#define HID     450
#define FIN     128
#define E2      (N_EDGES + N_NODES)
#define FC1_BLOCKS 675
#define FC1_ROWS   400      // 675*400 = 270000

__global__ void zero_kernel(float* __restrict__ p, int n) {
    int i = blockIdx.x * blockDim.x + threadIdx.x;
    if (i < n) p[i] = 0.f;
}

// ---- Fused NNConv: z = relu(ea@w1+b1); h = z@w2+b2 (WMMA); msg = x[src]·h; atomic scatter to agg[dst]
__global__ __launch_bounds__(128) void nnconv_kernel(
    const float* __restrict__ edge_attr, const int* __restrict__ src,
    const int* __restrict__ dst, const float* __restrict__ x,
    const float* __restrict__ w1, const float* __restrict__ b1,
    const float* __restrict__ w2, const float* __restrict__ b2,
    float* __restrict__ agg) {
    __shared__ _Float16 w2h[KDIM * HCOL];   // 20 KB, loaded once
    __shared__ float    b2s[HCOL];
    __shared__ _Float16 zh[16 * KDIM];      // A tile: 16 edges x K=32
    __shared__ float    xs[16 * F_IN];      // x[src] for the 16 edges
    __shared__ int      dsts[16];
    const int tid  = threadIdx.x;
    const int wave = tid >> 5;
    const int lane = tid & 31;

    for (int i = tid; i < KDIM * HCOL; i += 128) w2h[i] = (_Float16)w2[i];
    for (int i = tid; i < HCOL; i += 128) b2s[i] = b2[i];
    __syncthreads();

    // B fragments (32x16 f16, ISA layout: lanes 0-15 K=0..7,16..23, lanes 16-31 K=8..15,24..31),
    // built once; wave w owns column tiles T = t*4 + w  ->  col = t*64 + (16w + lane%16)
    const int nloc = lane & 15;
    const int kbB  = (lane < 16) ? 0 : 8;
    v16h bfrag[5];
    for (int t = 0; t < 5; ++t) {
        int col = t * 64 + wave * 16 + nloc;
        for (int j = 0; j < 16; ++j) bfrag[t][j] = w2h[(kbB + j + (j & 8)) * HCOL + col];
    }
    const int o   = wave * 16 + nloc;           // output feature owned by this lane
    const int kbA = (lane < 16) ? 0 : 8;        // A-matrix split-K layout
    const int numTiles = N_EDGES / 16;

    for (int tile = blockIdx.x; tile < numTiles; tile += gridDim.x) {
        const int e0 = tile * 16;
        __syncthreads();
        // z tile (relu(ea@w1+b1)) -> f16 in LDS
        for (int idx = tid; idx < 16 * KDIM; idx += 128) {
            int el = idx >> 5, k = idx & 31, e = e0 + el;
            float zv = edge_attr[e * 2 + 0] * w1[k] +
                       edge_attr[e * 2 + 1] * w1[KDIM + k] + b1[k];
            zh[idx] = (_Float16)fmaxf(zv, 0.f);
        }
        if (tid < 16 * F_IN) {
            int el = tid / F_IN, i = tid % F_IN;
            xs[tid] = x[src[e0 + el] * F_IN + i];
        }
        if (tid < 16) dsts[tid] = dst[e0 + tid];
        __syncthreads();

        // A fragment per ISA 16-bit 16x32 layout
        v16h a;
        for (int j = 0; j < 8; ++j) a[j]     = zh[nloc * KDIM + kbA + j];
        for (int j = 0; j < 8; ++j) a[8 + j] = zh[nloc * KDIM + kbA + 16 + j];

        v8f acc[5];
        for (int t = 0; t < 5; ++t) {
            v8f cz = {};
            acc[t] = __builtin_amdgcn_wmma_f32_16x16x32_f16(
                false, a, false, bfrag[t], (short)0, cz, false, false);
        }
        // contraction: msg[e][o] = sum_i xs[e][i]*(h[i*64+o]); lane owns fixed o across all i
        const int rbase = (lane < 16) ? 0 : 8;
        for (int r = 0; r < 8; ++r) {
            int el = rbase + r;
            float msg = 0.f;
            for (int t = 0; t < 5; ++t)
                msg += xs[el * F_IN + t] * (acc[t][r] + b2s[t * 64 + o]);
            atomicAdd(&agg[dsts[el] * F_OUT + o], msg);
        }
    }
}

// x1 = relu(agg + x@root + bias)
__global__ void node_update_kernel(const float* __restrict__ agg, const float* __restrict__ x,
                                   const float* __restrict__ root, const float* __restrict__ bias,
                                   float* __restrict__ x1) {
    int idx = blockIdx.x * blockDim.x + threadIdx.x;
    if (idx >= N_NODES * F_OUT) return;
    int n = idx / F_OUT, oo = idx % F_OUT;
    float s = agg[idx] + bias[oo];
    for (int i = 0; i < F_IN; ++i) s += x[n * F_IN + i] * root[i * F_OUT + oo];
    x1[idx] = fmaxf(s, 0.f);
}

// xl = x1@gat_w ; a_src/a_dst = xl·att
__global__ __launch_bounds__(32) void gat_prep_kernel(
    const float* __restrict__ x1, const float* __restrict__ gw,
    const float* __restrict__ att_s, const float* __restrict__ att_d,
    float* __restrict__ xl, float* __restrict__ asrc, float* __restrict__ adst) {
    __shared__ float xrow[F_OUT];
    __shared__ float xlr[C_GAT];
    int n = blockIdx.x, t = threadIdx.x;
    for (int k = t; k < F_OUT; k += 32) xrow[k] = x1[n * F_OUT + k];
    __syncthreads();
    if (t < C_GAT) {
        float a = 0.f;
        for (int k = 0; k < F_OUT; ++k) a += xrow[k] * gw[k * C_GAT + t];
        xl[n * C_GAT + t] = a;
        xlr[t] = a;
    }
    __syncthreads();
    if (t == 0) {
        float as = 0.f, ad = 0.f;
        for (int c = 0; c < C_GAT; ++c) { as += xlr[c] * att_s[c]; ad += xlr[c] * att_d[c]; }
        asrc[n] = as; adst[n] = ad;
    }
}

// softmax denominator (shift-invariant; self-loops guarantee denom > 0)
__global__ void gat_denom_kernel(const int* __restrict__ src, const int* __restrict__ dst,
                                 const float* __restrict__ asrc, const float* __restrict__ adst,
                                 float* __restrict__ exbuf, float* __restrict__ denom) {
    int e = blockIdx.x * blockDim.x + threadIdx.x;
    if (e >= E2) return;
    int s, d;
    if (e < N_EDGES) { s = src[e]; d = dst[e]; } else { s = d = e - N_EDGES; }
    float al = asrc[s] + adst[d];
    al = (al >= 0.f) ? al : 0.2f * al;
    float ex = expf(al);
    exbuf[e] = ex;
    atomicAdd(&denom[d], ex);
}

// weighted aggregation: out2[d] += (ex/denom[d]) * xl[s]
__global__ void gat_agg_kernel(const int* __restrict__ src, const int* __restrict__ dst,
                               const float* __restrict__ exbuf, const float* __restrict__ denom,
                               const float* __restrict__ xl, float* __restrict__ out2) {
    long idx = (long)blockIdx.x * blockDim.x + threadIdx.x;
    int e = (int)(idx >> 5), c = (int)(idx & 31);
    if (e >= E2 || c >= C_GAT) return;
    int s, d;
    if (e < N_EDGES) { s = src[e]; d = dst[e]; } else { s = d = e - N_EDGES; }
    float attn = exbuf[e] / denom[d];
    atomicAdd(&out2[d * C_GAT + c], attn * xl[s * C_GAT + c]);
}

__global__ void gat_relu_kernel(const float* __restrict__ out2, const float* __restrict__ gbias,
                                float* __restrict__ v) {
    int idx = blockIdx.x * blockDim.x + threadIdx.x;
    if (idx >= N_NODES * C_GAT) return;
    v[idx] = fmaxf(out2[idx] + gbias[idx % C_GAT], 0.f);
}

// fc1: stream 486 MB of weights, deterministic partials per block
__global__ __launch_bounds__(512) void fc1_partial_kernel(const float* __restrict__ v,
                                                          const float* __restrict__ w,
                                                          float* __restrict__ partial) {
    __shared__ float vs[FC1_ROWS];
    int b = blockIdx.x, t = threadIdx.x;
    int n0 = b * FC1_ROWS;
    for (int r = t; r < FC1_ROWS; r += 512) vs[r] = v[n0 + r];
    __syncthreads();
    if (t < HID) {
        float acc = 0.f;
        const float* wp = w + (long)n0 * HID + t;
        for (int r = 0; r < FC1_ROWS; ++r) acc += vs[r] * wp[(long)r * HID];
        partial[b * HID + t] = acc;
    }
}

// reduce partials + relu + fc2 + relu
__global__ __launch_bounds__(512) void head_kernel(const float* __restrict__ partial,
                                                   const float* __restrict__ fb1,
                                                   const float* __restrict__ w2,
                                                   const float* __restrict__ fb2,
                                                   float* __restrict__ out) {
    __shared__ float hid[HID];
    int t = threadIdx.x;
    if (t < HID) {
        float s = fb1[t];
        for (int b = 0; b < FC1_BLOCKS; ++b) s += partial[b * HID + t];
        hid[t] = fmaxf(s, 0.f);
    }
    __syncthreads();
    if (t < FIN) {
        float s = fb2[t];
        for (int k = 0; k < HID; ++k) s += hid[k] * w2[k * FIN + t];
        out[t] = fmaxf(s, 0.f);
    }
}

extern "C" void kernel_launch(void* const* d_in, const int* in_sizes, int n_in,
                              void* d_out, int out_size, void* d_ws, size_t ws_size,
                              hipStream_t stream) {
    (void)in_sizes; (void)n_in; (void)out_size; (void)ws_size;
    const float* x     = (const float*)d_in[0];
    const int*   ei    = (const int*)  d_in[1];
    const float* ea    = (const float*)d_in[2];
    const float* mw1   = (const float*)d_in[3];
    const float* mb1   = (const float*)d_in[4];
    const float* mw2   = (const float*)d_in[5];
    const float* mb2   = (const float*)d_in[6];
    const float* root  = (const float*)d_in[7];
    const float* nbias = (const float*)d_in[8];
    const float* gw    = (const float*)d_in[9];
    const float* atts  = (const float*)d_in[10];
    const float* attd  = (const float*)d_in[11];
    const float* gbias = (const float*)d_in[12];
    const float* f1w   = (const float*)d_in[13];
    const float* f1b   = (const float*)d_in[14];
    const float* f2w   = (const float*)d_in[15];
    const float* f2b   = (const float*)d_in[16];
    float* out = (float*)d_out;

    float* ws    = (float*)d_ws;
    float* agg   = ws;                 // N*64      = 640000
    float* denom = agg   + 640000;     // N         =  10000
    float* out2  = denom + 10000;      // N*27      = 270000   (zero region ends here)
    float* x1    = out2  + 270000;     // N*64
    float* xl    = x1    + 640000;     // N*27
    float* asrc  = xl    + 270000;     // N
    float* adst  = asrc  + 10000;      // N
    float* exb   = adst  + 10000;      // E2        = 330000
    float* vv    = exb   + 330000;     // N*27
    float* part  = vv    + 270000;     // 675*450   = 303750
    const int* srcI = ei;
    const int* dstI = ei + N_EDGES;

    const int zeroN = 640000 + 10000 + 270000;
    zero_kernel<<<(zeroN + 255) / 256, 256, 0, stream>>>(agg, zeroN);
    nnconv_kernel<<<640, 128, 0, stream>>>(ea, srcI, dstI, x, mw1, mb1, mw2, mb2, agg);
    node_update_kernel<<<(N_NODES * F_OUT + 255) / 256, 256, 0, stream>>>(agg, x, root, nbias, x1);
    gat_prep_kernel<<<N_NODES, 32, 0, stream>>>(x1, gw, atts, attd, xl, asrc, adst);
    gat_denom_kernel<<<(E2 + 255) / 256, 256, 0, stream>>>(srcI, dstI, asrc, adst, exb, denom);
    long t5 = (long)E2 * 32;
    gat_agg_kernel<<<(int)((t5 + 255) / 256), 256, 0, stream>>>(srcI, dstI, exb, denom, xl, out2);
    gat_relu_kernel<<<(N_NODES * C_GAT + 255) / 256, 256, 0, stream>>>(out2, gbias, vv);
    fc1_partial_kernel<<<FC1_BLOCKS, 512, 0, stream>>>(vv, f1w, part);
    head_kernel<<<1, 512, 0, stream>>>(part, f1b, f2w, f2b, out);
}
